// EnhancedFractalTransformerBlock_43112881717573
// MI455X (gfx1250) — compile-verified, matrix operand order checked by reading, not run on hardware
//
#include <hip/hip_runtime.h>
#include <hip/hip_bf16.h>

// ---------------- constants ----------------
#define BB 4
#define SS 1024
#define DD 512
#define HH 8
#define DHH 64
#define MLPC 2048
#define MLV 50
#define S2 (SS * SS)

typedef __attribute__((ext_vector_type(16))) _Float16 v16h;
typedef __attribute__((ext_vector_type(8)))  float    v8f;

struct alignas(16) Q4 { unsigned int x, y, z, w; };

__device__ __forceinline__ v16h load_frag(const _Float16* p) {
    // 16-bit operand fragment: elements 0..7 = K kb..kb+7, 8..15 = kb+16..kb+23
    union { v16h v; struct { Q4 lo, hi; } q; } u;
    u.q.lo = *reinterpret_cast<const Q4*>(p);
    u.q.hi = *reinterpret_cast<const Q4*>(p + 16);
    return u.v;
}

// ---------------- generic f16 WMMA GEMM: C[M,N] = A[M,K] @ W[N,K]^T (+bias[n]) ----------------
// grid = (N/64, M/128), block = 256 (8 waves). Each wave: 32x32 tile (2x2 wmma accumulators).
__global__ void __launch_bounds__(256)
gemm_f16_wmma(const _Float16* __restrict__ A, int lda,
              const _Float16* __restrict__ W, int ldb,
              float* __restrict__ C, int ldc,
              const float* __restrict__ bias, int K) {
    const int lane  = threadIdx.x & 31;
    const int wave  = threadIdx.x >> 5;
    const int wm    = wave >> 1;       // 0..3
    const int wn    = wave & 1;        // 0..1
    const int row0  = blockIdx.y * 128 + wm * 32;
    const int col0  = blockIdx.x * 64  + wn * 32;
    const int fr    = lane & 15;
    const int khalf = (lane >> 4) << 3;   // 0 or 8

    v8f acc00 = {}, acc01 = {}, acc10 = {}, acc11 = {};
    for (int k = 0; k < K; k += 32) {
        const _Float16* pa0 = A + (size_t)(row0 + fr) * lda + (k + khalf);
        const _Float16* pa1 = pa0 + (size_t)16 * lda;
        const _Float16* pb0 = W + (size_t)(col0 + fr) * ldb + (k + khalf);
        const _Float16* pb1 = pb0 + (size_t)16 * ldb;
        v16h a0 = load_frag(pa0);
        v16h a1 = load_frag(pa1);
        v16h b0 = load_frag(pb0);
        v16h b1 = load_frag(pb1);
        acc00 = __builtin_amdgcn_wmma_f32_16x16x32_f16(false, a0, false, b0, (short)0, acc00, false, false);
        acc01 = __builtin_amdgcn_wmma_f32_16x16x32_f16(false, a0, false, b1, (short)0, acc01, false, false);
        acc10 = __builtin_amdgcn_wmma_f32_16x16x32_f16(false, a1, false, b0, (short)0, acc10, false, false);
        acc11 = __builtin_amdgcn_wmma_f32_16x16x32_f16(false, a1, false, b1, (short)0, acc11, false, false);
    }
    // C/D layout: VGPR r, lanes 0-15 -> M=r, lanes 16-31 -> M=8+r; N = lane&15
    const int moff = (lane >> 4) << 3;
    const int ncol = lane & 15;
    #pragma unroll
    for (int r = 0; r < 8; ++r) {
        int m0 = row0 + moff + r;
        int m1 = m0 + 16;
        int n0 = col0 + ncol;
        int n1 = n0 + 16;
        float bb0 = bias ? bias[n0] : 0.f;
        float bb1 = bias ? bias[n1] : 0.f;
        C[(size_t)m0 * ldc + n0] = acc00[r] + bb0;
        C[(size_t)m0 * ldc + n1] = acc01[r] + bb1;
        C[(size_t)m1 * ldc + n0] = acc10[r] + bb0;
        C[(size_t)m1 * ldc + n1] = acc11[r] + bb1;
    }
}

// ---------------- prep: depths, paths (fp32), path norms ----------------
__global__ void prep_kernel(const int* __restrict__ levels, int* __restrict__ depths,
                            float* __restrict__ paths, float* __restrict__ pn) {
    int s = blockIdx.x * 256 + threadIdx.x;
    if (s >= SS) return;
    int d = levels[s * 9];
    depths[s] = d < 0 ? 0 : (d > MLV ? MLV : d);
    float n2 = 0.f;
    #pragma unroll
    for (int p = 0; p < 8; ++p) {
        float v = (float)levels[s * 9 + 1 + p];
        paths[s * 8 + p] = v;
        n2 += v * v;
    }
    pn[s] = fmaxf(sqrtf(n2), 1e-8f);
}

// ---------------- hilbert-MLP + rel-pos bias, batch-invariant: bias[H,S,S] ----------------
__global__ void __launch_bounds__(256)
bias_kernel(const float* __restrict__ paths, const float* __restrict__ pn,
            const int* __restrict__ depths,
            const float* __restrict__ w1, const float* __restrict__ b1,
            const float* __restrict__ w2, const float* __restrict__ b2,
            const float* __restrict__ rel, float* __restrict__ biasOut) {
    int pair = blockIdx.x * 256 + threadIdx.x;
    int i = pair >> 10, j = pair & 1023;
    float d2 = 0.f, dot = 0.f;
    #pragma unroll
    for (int p = 0; p < 8; ++p) {
        float a = paths[i * 8 + p], b = paths[j * 8 + p];
        float df = a - b;
        d2 += df * df;
        dot += a * b;
    }
    float dist = sqrtf(d2);
    float sim = dot / (pn[i] * pn[j]);
    float hid[64];
    #pragma unroll
    for (int k = 0; k < 64; ++k)
        hid[k] = fmaxf(w1[2 * k] * dist + w1[2 * k + 1] * sim + b1[k], 0.f);
    int ld = depths[j] - depths[i];
    ld = ld < -MLV ? -MLV : (ld > MLV ? MLV : ld);
    ld += MLV;
    #pragma unroll
    for (int h = 0; h < HH; ++h) {
        float acc = b2[h];
        #pragma unroll
        for (int k = 0; k < 64; ++k) acc += w2[h * 64 + k] * hid[k];
        float hb = tanhf(acc);
        if (i == j) hb = 0.f;
        biasOut[((size_t)h << 20) + ((size_t)i << 10) + j] = 0.1f * hb + 0.05f * rel[ld * 8 + h];
    }
}

// ---------------- fused level-aware LN -> plain LN -> f16 ----------------
__global__ void __launch_bounds__(256)
ln_law_kernel(const float* __restrict__ xin,
              const float* __restrict__ gt, const float* __restrict__ bt,
              const int* __restrict__ depths,
              const float* __restrict__ g2, const float* __restrict__ b2,
              _Float16* __restrict__ outh) {
    __shared__ float sred[512];
    int token = blockIdx.x;           // b*S + s
    int s = token & (SS - 1);
    int dep = depths[s];
    const float* row = xin + (size_t)token * DD;
    int t = threadIdx.x;
    float v0 = row[t], v1 = row[t + 256];
    sred[t] = v0 + v1;
    sred[t + 256] = v0 * v0 + v1 * v1;
    __syncthreads();
    for (int st = 128; st; st >>= 1) {
        if (t < st) { sred[t] += sred[t + st]; sred[256 + t] += sred[256 + t + st]; }
        __syncthreads();
    }
    float m = sred[0] * (1.f / DD);
    float var = sred[256] * (1.f / DD) - m * m;
    __syncthreads();
    float inv = rsqrtf(var + 1e-5f);
    const float* g1r = gt + (size_t)dep * DD;
    const float* b1r = bt + (size_t)dep * DD;
    float y0 = (v0 - m) * inv * g1r[t] + b1r[t];
    float y1 = (v1 - m) * inv * g1r[t + 256] + b1r[t + 256];
    sred[t] = y0 + y1;
    sred[t + 256] = y0 * y0 + y1 * y1;
    __syncthreads();
    for (int st = 128; st; st >>= 1) {
        if (t < st) { sred[t] += sred[t + st]; sred[256 + t] += sred[256 + t + st]; }
        __syncthreads();
    }
    float m2 = sred[0] * (1.f / DD);
    float var2 = sred[256] * (1.f / DD) - m2 * m2;
    float inv2 = rsqrtf(var2 + 1e-5f);
    _Float16* orow = outh + (size_t)token * DD;
    orow[t]       = (_Float16)((y0 - m2) * inv2 * g2[t] + b2[t]);
    orow[t + 256] = (_Float16)((y1 - m2) * inv2 * g2[t + 256] + b2[t + 256]);
}

// ---------------- split qkv -> q_h, k_h (row-major [S,DH]) and vT ([DH,S]) per (b,h) ----------------
__global__ void __launch_bounds__(256)
qkv_split_kernel(const float* __restrict__ qkv, _Float16* __restrict__ qh,
                 _Float16* __restrict__ kh, _Float16* __restrict__ vT) {
    int idx = blockIdx.x * 256 + threadIdx.x;     // (b,h,s,d)
    int d = idx & 63;
    int s = (idx >> 6) & 1023;
    int h = (idx >> 16) & 7;
    int b = idx >> 19;
    const float* row = qkv + ((size_t)(b << 10) + s) * (3 * DD) + h * 64 + d;
    size_t bh = (size_t)(b * HH + h);
    qh[(bh * SS + s) * 64 + d] = (_Float16)row[0];
    kh[(bh * SS + s) * 64 + d] = (_Float16)row[DD];
    vT[(bh * 64 + d) * SS + s] = (_Float16)row[2 * DD];
}

// ---------------- fused scale + bias + softmax -> f16 probs (one (h,i) row per block) ----------------
__global__ void __launch_bounds__(256)
softmax_kernel(const float* __restrict__ scores, const float* __restrict__ biasIn,
               _Float16* __restrict__ attn, const float* __restrict__ scale_w,
               const float* __restrict__ lse, const int* __restrict__ depths) {
    __shared__ float sred[256];
    int h = blockIdx.x >> 10;
    int i = blockIdx.x & 1023;
    size_t rowoff = ((size_t)h << 20) + ((size_t)i << 10);
    const float* srow = scores + rowoff;
    const float* brow = biasIn + rowoff;
    _Float16* arow = attn + rowoff;
    float mult = 0.125f * scale_w[h] * lse[depths[i] * HH + h];   // 1/sqrt(64) = 0.125
    int t = threadIdx.x;
    float v[4];
    float mx = -3.4e38f;
    #pragma unroll
    for (int q = 0; q < 4; ++q) {
        v[q] = srow[t + q * 256] * mult + brow[t + q * 256];
        mx = fmaxf(mx, v[q]);
    }
    sred[t] = mx; __syncthreads();
    for (int st = 128; st; st >>= 1) { if (t < st) sred[t] = fmaxf(sred[t], sred[t + st]); __syncthreads(); }
    mx = sred[0]; __syncthreads();
    float sum = 0.f;
    #pragma unroll
    for (int q = 0; q < 4; ++q) { v[q] = expf(v[q] - mx); sum += v[q]; }
    sred[t] = sum; __syncthreads();
    for (int st = 128; st; st >>= 1) { if (t < st) sred[t] += sred[t + st]; __syncthreads(); }
    float inv = 1.f / sred[0];
    #pragma unroll
    for (int q = 0; q < 4; ++q) arow[t + q * 256] = (_Float16)(v[q] * inv);
}

// ---------------- elementwise helpers ----------------
__global__ void cvt_h_kernel(const float* __restrict__ src, _Float16* __restrict__ dst, int n) {
    int i = blockIdx.x * 256 + threadIdx.x;
    if (i < n) dst[i] = (_Float16)src[i];
}
__global__ void relu_cvt_kernel(const float* __restrict__ src, _Float16* __restrict__ dst, int n) {
    int i = blockIdx.x * 256 + threadIdx.x;
    if (i < n) dst[i] = (_Float16)fmaxf(src[i], 0.f);
}
__global__ void resid_kernel(const float* __restrict__ base, const float* __restrict__ addv,
                             const float* __restrict__ rw, int widx, float* __restrict__ outp, int n) {
    int i = blockIdx.x * 256 + threadIdx.x;
    if (i < n) outp[i] = base[i] + rw[widx] * addv[i];
}
__global__ void gate_mul_kernel(float* __restrict__ inb, const float* __restrict__ g2, int n) {
    int i = blockIdx.x * 256 + threadIdx.x;
    if (i < n) inb[i] = inb[i] * (1.f / (1.f + expf(-g2[i])));
}
__global__ void act_kernel(const float* __restrict__ gated, const float* __restrict__ aw,
                           _Float16* __restrict__ acth, int n) {
    int i = blockIdx.x * 256 + threadIdx.x;
    if (i >= n) return;
    int row = i >> 11;   // /MLPC
    float g = gated[i];
    float a0 = aw[row * 3], a1 = aw[row * 3 + 1], a2 = aw[row * 3 + 2];
    float gelu = 0.5f * g * (1.f + erff(g * 0.70710678118f));
    float rel = fmaxf(g, 0.f);
    float sil = g / (1.f + expf(-g));
    acth[i] = (_Float16)(a0 * gelu + a1 * rel + a2 * sil);
}

// ---------------- activation-mixing weights: aw = softmax(xf @ actW^T + actB) [rows, 3] ----------------
__global__ void __launch_bounds__(128)
aw_kernel(const _Float16* __restrict__ xf, const float* __restrict__ actW,
          const float* __restrict__ actB, float* __restrict__ aw) {
    __shared__ float sred[3][128];
    int row = blockIdx.x, t = threadIdx.x;
    const _Float16* xr = xf + (size_t)row * DD;
    float s0 = 0.f, s1 = 0.f, s2 = 0.f;
    for (int d = t; d < DD; d += 128) {
        float x = (float)xr[d];
        s0 += x * actW[d];
        s1 += x * actW[DD + d];
        s2 += x * actW[2 * DD + d];
    }
    sred[0][t] = s0; sred[1][t] = s1; sred[2][t] = s2;
    __syncthreads();
    for (int st = 64; st; st >>= 1) {
        if (t < st) {
            sred[0][t] += sred[0][t + st];
            sred[1][t] += sred[1][t + st];
            sred[2][t] += sred[2][t + st];
        }
        __syncthreads();
    }
    if (t == 0) {
        float l0 = sred[0][0] + actB[0], l1 = sred[1][0] + actB[1], l2 = sred[2][0] + actB[2];
        float m = fmaxf(l0, fmaxf(l1, l2));
        float e0 = expf(l0 - m), e1 = expf(l1 - m), e2 = expf(l2 - m);
        float inv = 1.f / (e0 + e1 + e2);
        aw[row * 3 + 0] = e0 * inv;
        aw[row * 3 + 1] = e1 * inv;
        aw[row * 3 + 2] = e2 * inv;
    }
}

// ==================================================================================
extern "C" void kernel_launch(void* const* d_in, const int* in_sizes, int n_in,
                              void* d_out, int out_size, void* d_ws, size_t ws_size,
                              hipStream_t stream) {
    (void)in_sizes; (void)n_in; (void)out_size; (void)ws_size;
    // ---- inputs (setup_inputs order) ----
    const float* x          = (const float*)d_in[0];
    const int*   levels     = (const int*)  d_in[1];
    const float* ln1_g      = (const float*)d_in[2];
    const float* ln1_b      = (const float*)d_in[3];
    const float* ln2_g      = (const float*)d_in[4];
    const float* ln2_b      = (const float*)d_in[5];
    const float* attn_ln_g  = (const float*)d_in[6];
    const float* attn_ln_b  = (const float*)d_in[7];
    const float* Wqkv       = (const float*)d_in[8];
    const float* scale_w    = (const float*)d_in[9];
    const float* lse        = (const float*)d_in[10];
    const float* hb_W1      = (const float*)d_in[11];
    const float* hb_b1      = (const float*)d_in[12];
    const float* hb_W2      = (const float*)d_in[13];
    const float* hb_b2      = (const float*)d_in[14];
    const float* rel_pos    = (const float*)d_in[15];
    const float* Wo         = (const float*)d_in[16];
    const float* bo         = (const float*)d_in[17];
    const float* ff_ln_g    = (const float*)d_in[18];
    const float* ff_ln_b    = (const float*)d_in[19];
    const float* W_in       = (const float*)d_in[20];
    const float* b_in       = (const float*)d_in[21];
    const float* W_out      = (const float*)d_in[22];
    const float* b_out      = (const float*)d_in[23];
    const float* gate_W1    = (const float*)d_in[24];
    const float* gate_b1    = (const float*)d_in[25];
    const float* gate_W2    = (const float*)d_in[26];
    const float* gate_b2    = (const float*)d_in[27];
    const float* act_W      = (const float*)d_in[28];
    const float* act_b      = (const float*)d_in[29];
    const float* resid_w    = (const float*)d_in[30];
    float* out = (float*)d_out;

    // ---- workspace layout (bump allocator with phase reuse) ----
    char* base = (char*)d_ws;
    size_t off = 0;
    auto alloc = [&](size_t bytes) { size_t o = off; off += (bytes + 255) & ~(size_t)255; return o; };
    const int NT = BB * SS;                 // 4096 tokens
    size_t depths_o = alloc(SS * 4);
    size_t paths_o  = alloc(SS * 8 * 4);
    size_t pn_o     = alloc(SS * 4);
    size_t wqkv_o   = alloc((size_t)3 * DD * DD * 2);        // [1536,512] f16
    size_t wo_o     = alloc((size_t)DD * DD * 2);
    size_t win_o    = alloc((size_t)MLPC * DD * 2);
    size_t wout_o   = alloc((size_t)DD * MLPC * 2);
    size_t gw1_o    = alloc((size_t)DD * DD * 2);            // MLP/4 = 512
    size_t gw2_o    = alloc((size_t)MLPC * DD * 2);
    size_t x2_o     = alloc((size_t)NT * DD * 4);
    size_t actin_o  = alloc((size_t)NT * DD * 2);            // xa_h then xf_h
    size_t o_o      = alloc((size_t)NT * DD * 4);
    size_t oh_o     = alloc((size_t)NT * DD * 2);            // o_h, later g1h
    size_t buf1_o   = alloc((size_t)HH * S2 * 4);            // bias -> inb
    size_t buf2_o   = alloc((size_t)HH * S2 * 4);            // scores -> attn_out -> g2
    size_t buf3_o   = alloc((size_t)NT * 3 * DD * 4);        // qkv -> g1 + ff + aw
    size_t buf4_o   = alloc((size_t)HH * S2 * 2);            // attn_h -> act_h
    size_t buf5_o   = alloc((size_t)3 * BB * HH * SS * DHH * 2); // q_h, k_h, vT

    int*      depths = (int*)(base + depths_o);
    float*    paths  = (float*)(base + paths_o);
    float*    pn     = (float*)(base + pn_o);
    _Float16* wqkv_h = (_Float16*)(base + wqkv_o);
    _Float16* wo_h   = (_Float16*)(base + wo_o);
    _Float16* win_h  = (_Float16*)(base + win_o);
    _Float16* wout_h = (_Float16*)(base + wout_o);
    _Float16* gw1_h  = (_Float16*)(base + gw1_o);
    _Float16* gw2_h  = (_Float16*)(base + gw2_o);
    float*    x2     = (float*)(base + x2_o);
    _Float16* actin  = (_Float16*)(base + actin_o);
    float*    obuf   = (float*)(base + o_o);
    _Float16* oh     = (_Float16*)(base + oh_o);
    float*    biasB  = (float*)(base + buf1_o);
    float*    inb    = (float*)(base + buf1_o);
    float*    scores = (float*)(base + buf2_o);
    float*    attnout= (float*)(base + buf2_o);
    float*    g2     = (float*)(base + buf2_o);
    float*    qkv    = (float*)(base + buf3_o);
    float*    g1     = (float*)(base + buf3_o);
    float*    ffbuf  = (float*)(base + buf3_o + (size_t)NT * DD * 4);
    float*    aw     = (float*)(base + buf3_o + (size_t)2 * NT * DD * 4);
    _Float16* attn_h = (_Float16*)(base + buf4_o);
    _Float16* act_h  = (_Float16*)(base + buf4_o);
    _Float16* q_h    = (_Float16*)(base + buf5_o);
    _Float16* k_h    = (_Float16*)(base + buf5_o + (size_t)BB * HH * SS * DHH * 2);
    _Float16* vT     = (_Float16*)(base + buf5_o + (size_t)2 * BB * HH * SS * DHH * 2);
    _Float16* g1h    = oh;   // reuse after o-proj

    const int nTok = NT * DD;        // 2M
    const int nBig = NT * MLPC;      // 8M
    auto blk = [](int n) { return (n + 255) / 256; };

    // ---- prep + batch-invariant bias ----
    prep_kernel<<<blk(SS), 256, 0, stream>>>(levels, depths, paths, pn);
    bias_kernel<<<S2 / 256, 256, 0, stream>>>(paths, pn, depths, hb_W1, hb_b1, hb_W2, hb_b2,
                                              rel_pos, biasB);
    // ---- weight conversions to f16 ----
    cvt_h_kernel<<<blk(3 * DD * DD), 256, 0, stream>>>(Wqkv, wqkv_h, 3 * DD * DD);
    cvt_h_kernel<<<blk(DD * DD), 256, 0, stream>>>(Wo, wo_h, DD * DD);
    cvt_h_kernel<<<blk(MLPC * DD), 256, 0, stream>>>(W_in, win_h, MLPC * DD);
    cvt_h_kernel<<<blk(DD * MLPC), 256, 0, stream>>>(W_out, wout_h, DD * MLPC);
    cvt_h_kernel<<<blk(DD * DD), 256, 0, stream>>>(gate_W1, gw1_h, DD * DD);
    cvt_h_kernel<<<blk(MLPC * DD), 256, 0, stream>>>(gate_W2, gw2_h, MLPC * DD);

    // ---- attention branch ----
    ln_law_kernel<<<NT, 256, 0, stream>>>(x, ln1_g, ln1_b, depths, attn_ln_g, attn_ln_b, actin);
    gemm_f16_wmma<<<dim3((3 * DD) / 64, NT / 128), 256, 0, stream>>>(
        actin, DD, wqkv_h, DD, qkv, 3 * DD, nullptr, DD);
    qkv_split_kernel<<<blk(BB * HH * SS * DHH), 256, 0, stream>>>(qkv, q_h, k_h, vT);

    for (int b = 0; b < BB; ++b) {
        for (int h = 0; h < HH; ++h) {
            size_t bh = (size_t)(b * HH + h);
            gemm_f16_wmma<<<dim3(SS / 64, SS / 128), 256, 0, stream>>>(
                q_h + bh * SS * DHH, DHH, k_h + bh * SS * DHH, DHH,
                scores + (size_t)h * S2, SS, nullptr, DHH);
        }
        softmax_kernel<<<HH * SS, 256, 0, stream>>>(scores, biasB, attn_h, scale_w, lse, depths);
        for (int h = 0; h < HH; ++h) {
            size_t bh = (size_t)(b * HH + h);
            gemm_f16_wmma<<<dim3(DHH / 64, SS / 128), 256, 0, stream>>>(
                attn_h + (size_t)h * S2, SS, vT + bh * DHH * SS, SS,
                obuf + (size_t)b * SS * DD + h * DHH, DD, nullptr, SS);
        }
    }
    cvt_h_kernel<<<blk(nTok), 256, 0, stream>>>(obuf, oh, nTok);
    gemm_f16_wmma<<<dim3(DD / 64, NT / 128), 256, 0, stream>>>(
        oh, DD, wo_h, DD, attnout, DD, bo, DD);
    resid_kernel<<<blk(nTok), 256, 0, stream>>>(x, attnout, resid_w, 0, x2, nTok);

    // ---- feed-forward branch ----
    ln_law_kernel<<<NT, 256, 0, stream>>>(x2, ln2_g, ln2_b, depths, ff_ln_g, ff_ln_b, actin);
    gemm_f16_wmma<<<dim3(DD / 64, NT / 128), 256, 0, stream>>>(
        actin, DD, gw1_h, DD, g1, DD, gate_b1, DD);
    relu_cvt_kernel<<<blk(nTok), 256, 0, stream>>>(g1, g1h, nTok);
    gemm_f16_wmma<<<dim3(MLPC / 64, NT / 128), 256, 0, stream>>>(
        g1h, DD, gw2_h, DD, g2, MLPC, gate_b2, DD);
    gemm_f16_wmma<<<dim3(MLPC / 64, NT / 128), 256, 0, stream>>>(
        actin, DD, win_h, DD, inb, MLPC, b_in, DD);
    gate_mul_kernel<<<blk(nBig), 256, 0, stream>>>(inb, g2, nBig);
    aw_kernel<<<NT, 128, 0, stream>>>(actin, act_W, act_b, aw);
    act_kernel<<<blk(nBig), 256, 0, stream>>>(inb, aw, act_h, nBig);
    gemm_f16_wmma<<<dim3(DD / 64, NT / 128), 256, 0, stream>>>(
        act_h, MLPC, wout_h, MLPC, ffbuf, DD, b_out, MLPC);
    resid_kernel<<<blk(nTok), 256, 0, stream>>>(x2, ffbuf, resid_w, 1, out, nTok);
}
